// MaskedNeuralRNN_50603304682320
// MI455X (gfx1250) — matrix operand order repaired and checked by reading, not compiled, not run on hardware
//
#include <hip/hip_runtime.h>
#include <hip/hip_bf16.h>
#include <math.h>

#define B_   64
#define S_   512
#define I_   512
#define H_   1024

// LDS B-tile row stride in bf16 elements: 32 data + 8 pad (80 bytes).
// Bank index for row n (16B loads) = (n*20) % 64 -> all-distinct for n=0..15.
#define BSTRIDE 40

typedef __attribute__((ext_vector_type(16))) __bf16 v16bf;
typedef __attribute__((ext_vector_type(8)))  __bf16 v8bf;
typedef __attribute__((ext_vector_type(8)))  float  v8f;
typedef __attribute__((ext_vector_type(8)))  float  f8;

union BF16x16 { v16bf v; v8bf h[2]; };

// ---------------- fragment loaders ----------------

// B fragment (32x16 KxN bf16): lane n=L&15, hi=L>>4 holds K = hi*16..hi*16+15 of
// column n. Weights are stored transposed, so this is 32 contiguous bytes.
__device__ __forceinline__ v16bf load_b_frag_global(const __bf16* bt_row, int k0, int hi) {
  return *(const v16bf*)(bt_row + k0 + hi * 16);
}

// Same B fragment out of the LDS-staged tile (two explicitly 16B-aligned loads).
__device__ __forceinline__ v16bf load_b_frag_lds(const __bf16* sm, int nloc, int hi) {
  const __bf16* p = sm + nloc * BSTRIDE + hi * 16;
  BF16x16 u;
  u.h[0] = *(const v8bf*)(p);
  u.h[1] = *(const v8bf*)(p + 8);
  return u.v;
}

// A fragment (16x32 MxK bf16): lane m=L&15, hi=L>>4 holds
// K = hi*16..hi*16+15.
__device__ __forceinline__ v16bf load_a_frag_bf16(const __bf16* a_row, int k0, int hi) {
  BF16x16 u;
  u.h[0] = *(const v8bf*)(a_row + k0 + hi * 16);
  u.h[1] = *(const v8bf*)(a_row + k0 + hi * 16 + 8);
  return u.v;
}

// A fragment sourced from fp32 (x stays fp32 in HBM; convert in-register).
__device__ __forceinline__ v16bf load_a_frag_f32(const float* a_row, int k0, int hi) {
  f8 f0 = *(const f8*)(a_row + k0 + hi * 16);
  f8 f1 = *(const f8*)(a_row + k0 + hi * 16 + 8);
  BF16x16 u;
  u.h[0] = __builtin_convertvector(f0, v8bf);
  u.h[1] = __builtin_convertvector(f1, v8bf);
  return u.v;
}

// ---------------- async LDS staging (gfx1250 ASYNCcnt path) ----------------

// Stage a 128-row x 32-element bf16 B-tile (8 KB) into LDS.
// 256 threads x 16 B x 2 rounds. Tracked by ASYNCcnt.
__device__ __forceinline__ void stage_b_chunk_async(const __bf16* __restrict__ bt,
                                                    int n0, int k0, int ldb,
                                                    __bf16* sm, int tid) {
  const int r4 = tid >> 2;   // row 0..63
  const int c4 = tid & 3;    // which 16-byte chunk of the 64-byte row
#pragma unroll
  for (int half = 0; half < 2; ++half) {
    const int row = r4 + half * 64;
    const __bf16* gp = bt + (size_t)(n0 + row) * ldb + k0 + c4 * 8;
    // Generic LDS pointer: low 32 bits are the LDS byte offset (ISA 10.2).
    unsigned loff = (unsigned)(uintptr_t)(sm + row * BSTRIDE + c4 * 8);
    asm volatile("global_load_async_to_lds_b128 %0, %1, off"
                 :: "v"(loff), "v"(gp)
                 : "memory");
  }
}

__device__ __forceinline__ void wait_async_all() {
  asm volatile("s_wait_asynccnt 0x0" ::: "memory");
}

// ---------------- kernels ----------------

// Mask + cast + transpose both weight matrices into bf16; zero the h0 buffer.
__global__ void prep_kernel(const float* __restrict__ Wxh, const int* __restrict__ maskx,
                            const float* __restrict__ Whh, const int* __restrict__ maskh,
                            __bf16* __restrict__ wxT, __bf16* __restrict__ whT,
                            __bf16* __restrict__ h0buf) {
  int t = blockIdx.x * blockDim.x + threadIdx.x;
  if (t < I_ * H_) {
    int i = t / H_, h = t % H_;
    wxT[h * I_ + i] = (__bf16)(Wxh[t] * (float)maskx[t]);
  }
  if (t < H_ * H_) {
    int h1 = t / H_, h2 = t % H_;
    whT[h2 * H_ + h1] = (__bf16)(Whh[t] * (float)maskh[t]);
  }
  if (t < B_ * H_) h0buf[t] = (__bf16)0.0f;
}

// Phase 1: hseq(=xproj) = x @ Wx + b_xh.  M=32768, K=512, N=1024.
// Block = 256 threads = 8 waves; block tile 128(M) x 128(N).
// B-tile double-buffered in LDS via async global->LDS copies.
__global__ __launch_bounds__(256) void xproj_kernel(const float* __restrict__ x,
                                                    const __bf16* __restrict__ wxT,
                                                    const float* __restrict__ bxh,
                                                    float* __restrict__ hseq) {
  __shared__ __bf16 bsm[2][128 * BSTRIDE];

  const int tid  = threadIdx.x;
  const int lane = tid & 31;
  const int wave = tid >> 5;
  const int hi   = lane >> 4;
  const int ln   = lane & 15;
  const int m0   = blockIdx.x * 128 + wave * 16;
  const int n0   = blockIdx.y * 128;

  const float* arow = x + (size_t)(m0 + ln) * I_;

  // Prime the pipeline with chunk 0.
  stage_b_chunk_async(wxT, n0, 0, I_, &bsm[0][0], tid);

  v8f c[8] = {};
  const int KC = I_ / 32;
  for (int kc = 0; kc < KC; ++kc) {
    const int k0 = kc * 32;
    wait_async_all();      // my chunk-kc stores have landed in LDS
    __syncthreads();       // everyone's have; everyone done reading buf[(kc+1)&1] (prev use)
    if (kc + 1 < KC)
      stage_b_chunk_async(wxT, n0, k0 + 32, I_, &bsm[(kc + 1) & 1][0], tid);

    v16bf a = load_a_frag_f32(arow, k0, hi);
    const __bf16* smc = &bsm[kc & 1][0];
    v16bf bf[8];
#pragma unroll
    for (int nt = 0; nt < 8; ++nt)
      bf[nt] = load_b_frag_lds(smc, nt * 16 + ln, hi);
#pragma unroll
    for (int nt = 0; nt < 8; ++nt)
      c[nt] = __builtin_amdgcn_wmma_f32_16x16x32_bf16(false, a, false, bf[nt],
                                                      (short)0, c[nt], false, false);
  }

#pragma unroll
  for (int nt = 0; nt < 8; ++nt) {
    const int col = n0 + nt * 16 + ln;
    const float bias = bxh[col];
#pragma unroll
    for (int r = 0; r < 8; ++r) {
      const int row = m0 + hi * 8 + r;             // C[hi*8+r][n] lives in c[r]
      hseq[(size_t)row * H_ + col] = c[nt][r] + bias;
    }
  }
}

// Phase 2 (one launch per timestep):
// h_t = tanh(h_{t-1} @ Wh + b_hh + xproj_t); xproj read from & h written to hseq.
// Wh^T (2 MB bf16) is L2-resident across the 512 launches -> direct global loads.
__global__ __launch_bounds__(256) void rnn_step_kernel(float* __restrict__ hseq,
                                                       const __bf16* __restrict__ whT,
                                                       const float* __restrict__ bhh,
                                                       const __bf16* __restrict__ hprev,
                                                       __bf16* __restrict__ hnext,
                                                       float* __restrict__ hfinal,
                                                       int t, int last) {
  const int lane = threadIdx.x & 31;
  const int wave = threadIdx.x >> 5;
  const int hi   = lane >> 4;
  const int ln   = lane & 15;
  const int m0   = (wave & 3) * 16;                    // batch-row strip
  const int n0   = blockIdx.x * 64 + (wave >> 2) * 32; // output-column strip

  const __bf16* arow  = hprev + (size_t)(m0 + ln) * H_;
  const __bf16* brow0 = whT + (size_t)(n0 + ln) * H_;
  const __bf16* brow1 = whT + (size_t)(n0 + 16 + ln) * H_;

  v8f c[2] = {};
  for (int k0 = 0; k0 < H_; k0 += 32) {
    __builtin_prefetch(arow + k0 + 128, 0, 0);   // global_prefetch_b8 on streamed A
    v16bf a  = load_a_frag_bf16(arow, k0, hi);
    v16bf b0 = load_b_frag_global(brow0, k0, hi);
    v16bf b1 = load_b_frag_global(brow1, k0, hi);
    c[0] = __builtin_amdgcn_wmma_f32_16x16x32_bf16(false, a, false, b0,
                                                   (short)0, c[0], false, false);
    c[1] = __builtin_amdgcn_wmma_f32_16x16x32_bf16(false, a, false, b1,
                                                   (short)0, c[1], false, false);
  }

#pragma unroll
  for (int nt = 0; nt < 2; ++nt) {
    const int col = n0 + nt * 16 + ln;
    const float bias = bhh[col];
#pragma unroll
    for (int r = 0; r < 8; ++r) {
      const int brow = m0 + hi * 8 + r;                       // batch index
      const size_t idx = ((size_t)brow * S_ + t) * H_ + col;  // h_seq[b][t][col]
      const float y = tanhf(c[nt][r] + bias + hseq[idx]);
      hseq[idx] = y;
      hnext[(size_t)brow * H_ + col] = (__bf16)y;
      if (last) hfinal[(size_t)brow * H_ + col] = y;
    }
  }
}

extern "C" void kernel_launch(void* const* d_in, const int* in_sizes, int n_in,
                              void* d_out, int out_size, void* d_ws, size_t ws_size,
                              hipStream_t stream) {
  const float* x     = (const float*)d_in[0];
  const float* Wxh   = (const float*)d_in[1];
  const float* bxh   = (const float*)d_in[2];
  const float* Whh   = (const float*)d_in[3];
  const float* bhh   = (const float*)d_in[4];
  const int*   maskx = (const int*)d_in[5];
  const int*   maskh = (const int*)d_in[6];

  float* hseq   = (float*)d_out;                     // (B,S,H)
  float* hfinal = hseq + (size_t)B_ * S_ * H_;       // (B,H)

  char* ws = (char*)d_ws;
  __bf16* wxT   = (__bf16*)ws;                                         // H*I bf16 (1 MB)
  __bf16* whT   = (__bf16*)(ws + (size_t)H_ * I_ * 2);                 // H*H bf16 (2 MB)
  __bf16* hbuf0 = (__bf16*)(ws + (size_t)H_ * I_ * 2 + (size_t)H_ * H_ * 2);
  __bf16* hbuf1 = hbuf0 + (size_t)B_ * H_;                             // 2x 128 KB ping-pong

  prep_kernel<<<(H_ * H_ + 255) / 256, 256, 0, stream>>>(Wxh, maskx, Whh, maskh,
                                                         wxT, whT, hbuf0);

  dim3 g1((B_ * S_) / 128, H_ / 128);
  xproj_kernel<<<g1, 256, 0, stream>>>(x, wxT, bxh, hseq);

  for (int t = 0; t < S_; ++t) {
    const __bf16* hp = (t & 1) ? hbuf1 : hbuf0;
    __bf16*       hn = (t & 1) ? hbuf0 : hbuf1;
    rnn_step_kernel<<<H_ / 64, 256, 0, stream>>>(hseq, whT, bhh, hp, hn, hfinal,
                                                 t, (t == S_ - 1) ? 1 : 0);
  }
}
